// PocketAffinityGuidance_88218628260297
// MI455X (gfx1250) — matrix operand order repaired and checked by reading, not compiled, not run on hardware
//
#include <hip/hip_runtime.h>
#include <hip/hip_bf16.h>

typedef __attribute__((ext_vector_type(16))) _Float16 v16h;
typedef __attribute__((ext_vector_type(8)))  float    v8f;

#define HD     256
#define BATCH  128
#define PPN    1024
#define LPN    128
#define NPROT  (BATCH * PPN)   // 131072
#define NLIG   (BATCH * LPN)   // 16384
#define NROWS  (NPROT + NLIG)  // 147456
#define RPB    64              // rows per block
#define KCH    8               // 256 / 32 k-chunks
#define CT     16              // 256 / 16 col tiles
#define FRAG_BYTES (32 * 32)   // one 32x16 f16 B fragment = 1024 bytes

// ---------------------------------------------------------------------------
// Pack a row-major [256 x 256] f32 weight matrix into the per-lane B-fragment
// layout this code uses for V_WMMA_F32_16X16X32_F16 (f16). Fragment f = ct*KCH + kc holds the
// 32x16 block  K = [32*kc, 32*kc+32), N = [16*ct, 16*ct+16).
// Lane l owns column n = 16*ct + (l%16); its 16 contiguous halves are
// K = 32*kc + 16*(l/16) + [0..15].
// ---------------------------------------------------------------------------
__global__ void pack_weights_kernel(const float* __restrict__ W,
                                    _Float16* __restrict__ Wp) {
  const int f  = blockIdx.x;          // 0..127
  const int ct = f / KCH;
  const int kc = f % KCH;
  const int l  = threadIdx.x;         // 0..31
  const int n     = ct * 16 + (l & 15);
  const int kbase = kc * 32 + 16 * (l >> 4);
  _Float16* dst = Wp + ((size_t)f * 32 + l) * 16;
#pragma unroll
  for (int idx = 0; idx < 16; ++idx)
    dst[idx] = (_Float16)W[(kbase + idx) * HD + n];
}

// ---------------------------------------------------------------------------
// Fused: embedding gather -> silu(X@Wd1+b1) -> @Wd2+b2 -> per-batch pooled sum
// 64 rows per block, 128 threads (4 waves), each wave owns 4 column tiles and
// all 4 row-subtiles (B fragment reused 4x per load).
// ---------------------------------------------------------------------------
__global__ __launch_bounds__(128) void mlp_pool_kernel(
    const int* __restrict__ pelem, const int* __restrict__ paa,
    const int* __restrict__ pbb,   const int* __restrict__ ltyp,
    const float* __restrict__ Eelem, const float* __restrict__ Eaa,
    const float* __restrict__ Ebb,   const float* __restrict__ Elig,
    const _Float16* __restrict__ Wd1p, const float* __restrict__ bd1,
    const _Float16* __restrict__ Wd2p, const float* __restrict__ bd2,
    float* __restrict__ poolP, float* __restrict__ poolL) {
  __shared__ _Float16 ldsX[RPB * HD];   // 32 KB
  __shared__ _Float16 ldsY[RPB * HD];   // 32 KB

  const int row0 = blockIdx.x * RPB;
  const int tid  = threadIdx.x;

  // ---- embedding gather into LDS (f16) ----
  for (int i = tid; i < RPB * HD; i += 128) {
    const int r = i >> 8;
    const int c = i & 255;
    const int g = row0 + r;
    float v;
    if (g < NPROT) {
      v = Eelem[pelem[g] * HD + c] + Eaa[paa[g] * HD + c] +
          Ebb[pbb[g] * HD + c];
    } else {
      v = Elig[ltyp[g - NPROT] * HD + c];
    }
    ldsX[i] = (_Float16)v;
  }
  __syncthreads();

  const int wave = tid >> 5;
  const int lane = tid & 31;
  const int lrow = lane & 15;   // A-matrix row / D-matrix column
  const int lhi  = lane >> 4;

  // Per-lane byte base inside a column-tile's run of 8 fragments; kc advances
  // by a compile-time FRAG_BYTES so loads use immediate offsets.
  const size_t laneByte = (size_t)lane * 32;

  // ---- layer 1: Y = silu(X @ Wd1 + b1), written back to LDS as f16 ----
  for (int ctl = 0; ctl < 4; ++ctl) {
    const int ct = wave * 4 + ctl;
    const char* bbase =
        (const char*)Wd1p + (size_t)ct * KCH * FRAG_BYTES + laneByte;
    v8f acc[4] = {};
#pragma unroll
    for (int kc = 0; kc < KCH; ++kc) {
      v16h bfrag;
      const float4* bp = (const float4*)(bbase + kc * FRAG_BYTES);
      ((float4*)&bfrag)[0] = bp[0];
      ((float4*)&bfrag)[1] = bp[1];
      __builtin_prefetch((const void*)(bbase + (kc + 1) * FRAG_BYTES), 0, 3);
      const int k0 = kc * 32 + 16 * lhi;
#pragma unroll
      for (int rt = 0; rt < 4; ++rt) {
        v16h afrag;
        const float4* ap = (const float4*)(&ldsX[(rt * 16 + lrow) * HD + k0]);
        ((float4*)&afrag)[0] = ap[0];
        ((float4*)&afrag)[1] = ap[1];   // second K run at k0+8 halves
        acc[rt] = __builtin_amdgcn_wmma_f32_16x16x32_f16(
            false, afrag, false, bfrag, (short)0, acc[rt], false, false);
      }
    }
    const int col = ct * 16 + lrow;
    const float b1 = bd1[col];
#pragma unroll
    for (int rt = 0; rt < 4; ++rt) {
#pragma unroll
      for (int r = 0; r < 8; ++r) {
        const float x = acc[rt][r] + b1;
        const float s = x / (1.0f + __expf(-x));   // silu
        ldsY[(rt * 16 + r + 8 * lhi) * HD + col] = (_Float16)s;
      }
    }
  }
  __syncthreads();

  // ---- layer 2: Z = Y @ Wd2 + b2, reduced directly into batch pool ----
  float* pool;
  if (row0 < NPROT) {
    pool = poolP + (size_t)(row0 / PPN) * HD;
  } else {
    pool = poolL + (size_t)((row0 - NPROT) / LPN) * HD;
  }

  for (int ctl = 0; ctl < 4; ++ctl) {
    const int ct = wave * 4 + ctl;
    const char* bbase =
        (const char*)Wd2p + (size_t)ct * KCH * FRAG_BYTES + laneByte;
    v8f acc[4] = {};
#pragma unroll
    for (int kc = 0; kc < KCH; ++kc) {
      v16h bfrag;
      const float4* bp = (const float4*)(bbase + kc * FRAG_BYTES);
      ((float4*)&bfrag)[0] = bp[0];
      ((float4*)&bfrag)[1] = bp[1];
      const int k0 = kc * 32 + 16 * lhi;
#pragma unroll
      for (int rt = 0; rt < 4; ++rt) {
        v16h afrag;
        const float4* ap = (const float4*)(&ldsY[(rt * 16 + lrow) * HD + k0]);
        ((float4*)&afrag)[0] = ap[0];
        ((float4*)&afrag)[1] = ap[1];
        acc[rt] = __builtin_amdgcn_wmma_f32_16x16x32_f16(
            false, afrag, false, bfrag, (short)0, acc[rt], false, false);
      }
    }
    const int col = ct * 16 + lrow;
    const float b2 = bd2[col];
    float partial = 0.0f;
#pragma unroll
    for (int rt = 0; rt < 4; ++rt)
#pragma unroll
      for (int r = 0; r < 8; ++r) partial += acc[rt][r];
    // lane l covers 32 rows; partner lane l^16 covers the other 32 (same col)
    partial += __shfl_xor(partial, 16, 32);
    if (lhi == 0) atomicAdd(&pool[col], partial + 64.0f * b2);
  }
}

// ---------------------------------------------------------------------------
// Per-batch ligand->protein min distance, then mean/min over ligand atoms.
// One block per batch, 128 threads (one per ligand atom).
// ---------------------------------------------------------------------------
__global__ __launch_bounds__(128) void contact_kernel(
    const float* __restrict__ ppos, const float* __restrict__ lpos,
    float* __restrict__ contact) {
  __shared__ float sp[PPN * 3];   // 12 KB
  __shared__ float rsum[LPN];
  __shared__ float rmin[LPN];
  const int b = blockIdx.x;
  const int t = threadIdx.x;

  for (int i = t; i < PPN * 3; i += 128)
    sp[i] = ppos[(size_t)b * PPN * 3 + i];
  __syncthreads();

  const size_t li = ((size_t)b * LPN + t) * 3;
  const float lx = lpos[li + 0];
  const float ly = lpos[li + 1];
  const float lz = lpos[li + 2];
  float dmin = 3.4e38f;
  for (int p = 0; p < PPN; ++p) {
    const float dx = lx - sp[p * 3 + 0];
    const float dy = ly - sp[p * 3 + 1];
    const float dz = lz - sp[p * 3 + 2];
    dmin = fminf(dmin, dx * dx + dy * dy + dz * dz);
  }
  const float d = sqrtf(dmin);
  rsum[t] = d;
  rmin[t] = d;
  __syncthreads();
  for (int s = 64; s > 0; s >>= 1) {
    if (t < s) {
      rsum[t] += rsum[t + s];
      rmin[t] = fminf(rmin[t], rmin[t + s]);
    }
    __syncthreads();
  }
  if (t == 0) {
    contact[2 * b + 0] = rsum[0] * (1.0f / LPN);
    contact[2 * b + 1] = rmin[0];
  }
}

// ---------------------------------------------------------------------------
// graph_feat = [poolP/1024, poolL/128, contact] (514) -> silu(@Wa1+ba1) -> @Wa2+ba2
// One block per batch row, 256 threads (one per hidden unit).
// ---------------------------------------------------------------------------
__global__ __launch_bounds__(256) void affinity_kernel(
    const float* __restrict__ poolP, const float* __restrict__ poolL,
    const float* __restrict__ contact, const float* __restrict__ Wa1,
    const float* __restrict__ ba1, const float* __restrict__ Wa2,
    const float* __restrict__ ba2, float* __restrict__ out) {
  __shared__ float gf[2 * HD + 2];
  __shared__ float hred[HD];
  const int b = blockIdx.x;
  const int t = threadIdx.x;

  gf[t]      = poolP[(size_t)b * HD + t] * (1.0f / PPN);
  gf[HD + t] = poolL[(size_t)b * HD + t] * (1.0f / LPN);
  if (t < 2) gf[2 * HD + t] = contact[2 * b + t];
  __syncthreads();

  float a = ba1[t];
  for (int i = 0; i < 2 * HD + 2; ++i) a += gf[i] * Wa1[i * HD + t];
  const float s = a / (1.0f + __expf(-a));
  hred[t] = s * Wa2[t];
  __syncthreads();
  for (int st = 128; st > 0; st >>= 1) {
    if (t < st) hred[t] += hred[t + st];
    __syncthreads();
  }
  if (t == 0) out[b] = hred[0] + ba2[0];
}

// ---------------------------------------------------------------------------
extern "C" void kernel_launch(void* const* d_in, const int* in_sizes, int n_in,
                              void* d_out, int out_size, void* d_ws,
                              size_t ws_size, hipStream_t stream) {
  const float* ppos  = (const float*)d_in[0];
  const float* lpos  = (const float*)d_in[1];
  const int*   pelem = (const int*)d_in[2];
  const int*   paa   = (const int*)d_in[3];
  const int*   pbb   = (const int*)d_in[4];
  const int*   ltyp  = (const int*)d_in[5];
  // d_in[6], d_in[7]: batch ids (implied by contiguous layout, unused)
  const float* Eelem = (const float*)d_in[8];
  const float* Eaa   = (const float*)d_in[9];
  const float* Ebb   = (const float*)d_in[10];
  const float* Elig  = (const float*)d_in[11];
  const float* Wd1   = (const float*)d_in[12];
  const float* bd1   = (const float*)d_in[13];
  const float* Wd2   = (const float*)d_in[14];
  const float* bd2   = (const float*)d_in[15];
  const float* Wa1   = (const float*)d_in[16];
  const float* ba1   = (const float*)d_in[17];
  const float* Wa2   = (const float*)d_in[18];
  const float* ba2   = (const float*)d_in[19];

  char* ws = (char*)d_ws;
  _Float16* Wd1p    = (_Float16*)(ws + 0);
  _Float16* Wd2p    = (_Float16*)(ws + 131072);
  float*    poolP   = (float*)(ws + 262144);
  float*    poolL   = (float*)(ws + 393216);
  float*    contact = (float*)(ws + 524288);

  // zero the pooled accumulators every call (graph-capture safe)
  hipMemsetAsync(poolP, 0, 2 * BATCH * HD * sizeof(float), stream);

  pack_weights_kernel<<<CT * KCH, 32, 0, stream>>>(Wd1, Wd1p);
  pack_weights_kernel<<<CT * KCH, 32, 0, stream>>>(Wd2, Wd2p);

  mlp_pool_kernel<<<NROWS / RPB, 128, 0, stream>>>(
      pelem, paa, pbb, ltyp, Eelem, Eaa, Ebb, Elig, Wd1p, bd1, Wd2p, bd2,
      poolP, poolL);

  contact_kernel<<<BATCH, 128, 0, stream>>>(ppos, lpos, contact);

  affinity_kernel<<<BATCH, 256, 0, stream>>>(poolP, poolL, contact, Wa1, ba1,
                                             Wa2, ba2, (float*)d_out);
}
